// MLA_17042430230985
// MI455X (gfx1250) — matrix operand order repaired and checked by reading, not compiled, not run on hardware
//
#include <hip/hip_runtime.h>

// ---------------- problem config ----------------
constexpr int Bc = 4, Sc = 1024, IDc = 4096;
constexpr int Rc = 512, RPc = 64, Hc = 32, Dc = 128;
constexpr int BS = Bc * Sc;     // 4096 tokens
constexpr int HD = Hc * Dc;     // 4096
constexpr int RR = Rc + RPc;    // 576
constexpr float EPSF = 1.1920929e-07f;

typedef __bf16 bf_t;
typedef __attribute__((ext_vector_type(16))) __bf16 v16bf;
typedef __attribute__((ext_vector_type(8)))  __bf16 v8bf;
typedef __attribute__((ext_vector_type(8)))  float  v8f;

// ---------------- generic NT GEMM descriptor ----------------
// C[z][m][n] = sum_k A[z][m][k] * B[z][n][k]  (+bias[n]) (+add[z][m][n])
// batch offsets (elements): off(z) = (z >> sh)*hi + (z & ((1<<sh)-1))*lo
// (all batch divisors in this pipeline are powers of two: 32 or "none")
struct Gemm {
  const bf_t* A; const bf_t* B;
  float* Cf; bf_t* Cb;                 // exactly one non-null
  const float* bias; const float* add; // optional
  long long a_row, a_hi, a_lo;
  long long b_row, b_hi, b_lo;
  long long c_row, c_bhi, c_blo;
  long long add_row, add_bhi, add_blo;
  int a_sh, b_sh, c_bsh, add_bsh;
  int M, N, K;
};

__device__ __forceinline__ long long boff(long long z, int sh, long long hi, long long lo) {
  return (z >> sh) * hi + (z & ((1LL << sh) - 1)) * lo;
}

// 1 wave32 computes a 64x64 tile of C via a 4x4 grid of 16x16x32 bf16 WMMAs.
// Requires M%64==0, N%64==0, K%32==0 (true for every GEMM in this pipeline).
__global__ __launch_bounds__(32) void gemm_nt_wmma(Gemm g) {
#if defined(__HIP_DEVICE_COMPILE__)
  const long long z  = blockIdx.z;
  const long long m0 = (long long)blockIdx.y * 64;
  const long long n0 = (long long)blockIdx.x * 64;
  const int lane = (int)(threadIdx.x & 31);
  const int r16  = lane & 15;
  const int hi   = lane >> 4;

  const bf_t* A  = g.A + boff(z, g.a_sh, g.a_hi, g.a_lo);
  const bf_t* Bt = g.B + boff(z, g.b_sh, g.b_hi, g.b_lo);

  v8f acc[4][4];
#pragma unroll
  for (int i = 0; i < 4; ++i)
#pragma unroll
    for (int j = 0; j < 4; ++j)
#pragma unroll
      for (int v = 0; v < 8; ++v) acc[i][j][v] = 0.0f;

  for (int k0 = 0; k0 < g.K; k0 += 32) {
    v16bf af[4], bfr[4];
#pragma unroll
    for (int i = 0; i < 4; ++i) {
      // A fragment (16x32, ISA 7.12.2): lane<16 -> K = k0+0..7 & k0+16..23
      const bf_t* ap = A + (m0 + i * 16 + r16) * g.a_row + k0 + hi * 8;
      union { v16bf v; v8bf h[2]; } ua;
      ua.h[0] = *(const v8bf*)(ap);
      ua.h[1] = *(const v8bf*)(ap + 16);
      af[i] = ua.v;
      // B fragment (32x16): lane<16 holds col n, K = k0+0..7 & k0+16..23
      const bf_t* bp = Bt + (n0 + i * 16 + r16) * g.b_row + k0 + hi * 8;
      union { v16bf v; v8bf h[2]; } ub;
      ub.h[0] = *(const v8bf*)(bp);
      ub.h[1] = *(const v8bf*)(bp + 16);
      bfr[i] = ub.v;
      __builtin_prefetch(ap + 32, 0, 1);   // global_prefetch_b8
      __builtin_prefetch(bp + 32, 0, 1);
    }
#pragma unroll
    for (int i = 0; i < 4; ++i)
#pragma unroll
      for (int j = 0; j < 4; ++j)
        acc[i][j] = __builtin_amdgcn_wmma_f32_16x16x32_bf16(
            false, af[i], false, bfr[j], (short)0, acc[i][j], false, false);
  }

  const long long cb = boff(z, g.c_bsh, g.c_bhi, g.c_blo);
  long long ab = 0;
  if (g.add) ab = boff(z, g.add_bsh, g.add_bhi, g.add_blo);
#pragma unroll
  for (int j = 0; j < 4; ++j) {
    const long long col = n0 + j * 16 + r16;
    const float bsv = g.bias ? g.bias[col] : 0.0f;
#pragma unroll
    for (int i = 0; i < 4; ++i) {
#pragma unroll
      for (int v = 0; v < 8; ++v) {
        // C 16x16 layout: VGPR v -> M = v + 8*(lane>=16), N = lane&15
        const long long row = m0 + i * 16 + hi * 8 + v;
        float val = acc[i][j][v] + bsv;
        if (g.add) val += g.add[ab + row * g.add_row + col];
        const long long co = cb + row * g.c_row + col;
        if (g.Cf) g.Cf[co] = val;
        else      g.Cb[co] = (bf_t)val;
      }
    }
  }
#endif
}

// ---------------- fp32 -> bf16 convert ----------------
__global__ void f2b(const float* __restrict__ s, bf_t* __restrict__ d, long long n) {
  long long i = (long long)blockIdx.x * blockDim.x + threadIdx.x;
  if (i < n) d[i] = (bf_t)s[i];
}

// ---------------- fused RMSNorm(kv) + RoPE(k_r) + RMSNorm(q) ----------------
// Writes c_kv / k_r TRANSPOSED into ckvkrT[b][j][s] (j<512: c_kv row j; j>=512: k_r)
// so that every downstream GEMM is NT. Writes c_q row-major (GEMM A operand).
__global__ __launch_bounds__(256) void norm_rope(
    const float* __restrict__ dkv, const float* __restrict__ dq,
    const float* __restrict__ kvw, const float* __restrict__ qnw,
    const float* __restrict__ cosh, const float* __restrict__ sinh,
    bf_t* __restrict__ ckvkrT, bf_t* __restrict__ cq) {
  __shared__ float red0[8], red1[8];
  const int t = blockIdx.x;          // token in [0, B*S)
  const int b = t >> 10, s = t & (Sc - 1);
  const int tid = threadIdx.x;
  const float* dkvp = dkv + (long long)t * RR;
  const float* dqp  = dq  + (long long)t * Rc;
  float s1 = 0.f, s2 = 0.f;
  for (int j = tid; j < Rc; j += 256) {
    float a = dkvp[j]; s1 += a * a;
    float c = dqp[j];  s2 += c * c;
  }
  for (int o = 16; o; o >>= 1) { s1 += __shfl_down(s1, o); s2 += __shfl_down(s2, o); }
  if ((tid & 31) == 0) { red0[tid >> 5] = s1; red1[tid >> 5] = s2; }
  __syncthreads();
  float t1 = 0.f, t2 = 0.f;
  for (int w = 0; w < 8; ++w) { t1 += red0[w]; t2 += red1[w]; }
  const float r1 = rsqrtf(t1 / (float)Rc + EPSF);
  const float r2 = rsqrtf(t2 / (float)Rc + EPSF);
  for (int j = tid; j < Rc; j += 256) {
    ckvkrT[((long long)b * RR + j) * Sc + s] = (bf_t)(dkvp[j] * r1 * kvw[j]);
    cq[(long long)t * Rc + j] = (bf_t)(dqp[j] * r2 * qnw[j]);
  }
  if (tid < RPc) {  // RoPE on decoupled key, half/half pairing
    int j = tid;
    float xv  = dkvp[Rc + j];
    float rot = (j < 32) ? -dkvp[Rc + j + 32] : dkvp[Rc + j - 32];
    float c = cosh[s * 32 + (j & 31)], sn = sinh[s * 32 + (j & 31)];
    ckvkrT[((long long)b * RR + Rc + j) * Sc + s] = (bf_t)(xv * c + rot * sn);
  }
}

// ---------------- RoPE on q_r ----------------
__global__ void rope64(const float* __restrict__ in, const float* __restrict__ cosh,
                       const float* __restrict__ sinh, bf_t* __restrict__ out) {
  long long t = blockIdx.x; int j = threadIdx.x;
  int s = (int)(t & (Sc - 1));
  float xv  = in[t * RPc + j];
  float rot = (j < 32) ? -in[t * RPc + j + 32] : in[t * RPc + j - 32];
  out[t * RPc + j] = (bf_t)(xv * cosh[s * 32 + (j & 31)] + rot * sinh[s * 32 + (j & 31)]);
}

// ---------------- launch ----------------
extern "C" void kernel_launch(void* const* d_in, const int* in_sizes, int n_in,
                              void* d_out, int out_size, void* d_ws, size_t ws_size,
                              hipStream_t stream) {
  (void)in_sizes; (void)n_in; (void)out_size; (void)ws_size;
  const float* x    = (const float*)d_in[0];
  const float* ch   = (const float*)d_in[1];
  const float* sh   = (const float*)d_in[2];
  const float* mask = (const float*)d_in[3];
  const float* wdkv = (const float*)d_in[4];
  const float* bdkv = (const float*)d_in[5];
  const float* kvnw = (const float*)d_in[6];
  const float* wdq  = (const float*)d_in[7];
  const float* bdq  = (const float*)d_in[8];
  const float* qnw  = (const float*)d_in[9];
  const float* wuq  = (const float*)d_in[10];
  const float* buq  = (const float*)d_in[11];
  const float* wqr  = (const float*)d_in[12];
  const float* bqr  = (const float*)d_in[13];
  const float* wuk  = (const float*)d_in[14];
  const float* wuv  = (const float*)d_in[15];
  const float* wo   = (const float*)d_in[16];
  const float* bo   = (const float*)d_in[17];
  float* out = (float*)d_out;

  // ---- workspace carve (256B aligned) ----
  size_t off = 0;
  auto carve = [&](size_t bytes) -> void* {
    void* p = (void*)((char*)d_ws + off);
    off += (bytes + 255) & ~(size_t)255;
    return p;
  };
  bf_t* xb     = (bf_t*)carve((size_t)BS * IDc * 2);
  bf_t* wdkvb  = (bf_t*)carve((size_t)RR * IDc * 2);
  bf_t* wdqb   = (bf_t*)carve((size_t)Rc * IDc * 2);
  bf_t* wuqb   = (bf_t*)carve((size_t)HD * Rc * 2);
  bf_t* wqrb   = (bf_t*)carve((size_t)RPc * Rc * 2);
  bf_t* wukb   = (bf_t*)carve((size_t)HD * Rc * 2);
  bf_t* wuvb   = (bf_t*)carve((size_t)HD * Rc * 2);
  bf_t* wob    = (bf_t*)carve((size_t)IDc * HD * 2);
  bf_t* maskb  = (bf_t*)carve((size_t)Sc * Sc * 2);
  float* dkv   = (float*)carve((size_t)BS * RR * 4);
  float* dq    = (float*)carve((size_t)BS * Rc * 4);
  bf_t* cq     = (bf_t*)carve((size_t)BS * Rc * 2);
  bf_t* qb     = (bf_t*)carve((size_t)BS * HD * 2);
  float* qr0   = (float*)carve((size_t)BS * RPc * 4);
  bf_t* qrb    = (bf_t*)carve((size_t)BS * RPc * 2);
  bf_t* ckvkrT = (bf_t*)carve((size_t)Bc * RR * Sc * 2);      // [b][j<576][s]
  bf_t* KCt    = (bf_t*)carve((size_t)Bc * Rc * RR * 2);      // [b][r][j]
  float* MC    = (float*)carve((size_t)Bc * Sc * Rc * 4);     // mask@C + P
  bf_t* W2     = (bf_t*)carve((size_t)Bc * Hc * Rc * Dc * 2); // [b][h][r][d]
  bf_t* asum   = (bf_t*)carve((size_t)Bc * Hc * Sc * Rc * 2); // [b][h][s][r]
  bf_t* attnb  = (bf_t*)carve((size_t)BS * HD * 2);

  auto cvt = [&](const float* s, bf_t* d, long long n) {
    f2b<<<dim3((unsigned)((n + 255) / 256)), dim3(256), 0, stream>>>(s, d, n);
  };
  auto mkg = [&]() {
    Gemm g; g.A = nullptr; g.B = nullptr; g.Cf = nullptr; g.Cb = nullptr;
    g.bias = nullptr; g.add = nullptr;
    g.a_row = 0; g.a_hi = 0; g.a_lo = 0;
    g.b_row = 0; g.b_hi = 0; g.b_lo = 0;
    g.c_row = 0; g.c_bhi = 0; g.c_blo = 0;
    g.add_row = 0; g.add_bhi = 0; g.add_blo = 0;
    g.a_sh = 30; g.b_sh = 30; g.c_bsh = 30; g.add_bsh = 30;  // 30 = "unbatched": z>>30==0
    g.M = 0; g.N = 0; g.K = 0; return g;
  };
  auto run = [&](const Gemm& g, int batch) {
    gemm_nt_wmma<<<dim3((unsigned)(g.N / 64), (unsigned)(g.M / 64), (unsigned)batch),
                   dim3(32, 1, 1), 0, stream>>>(g);
  };

  // ---- bf16 staging of inputs ----
  cvt(x,    xb,    (long long)BS * IDc);
  cvt(wdkv, wdkvb, (long long)RR * IDc);
  cvt(wdq,  wdqb,  (long long)Rc * IDc);
  cvt(wuq,  wuqb,  (long long)HD * Rc);
  cvt(wqr,  wqrb,  (long long)RPc * Rc);
  cvt(wuk,  wukb,  (long long)HD * Rc);
  cvt(wuv,  wuvb,  (long long)HD * Rc);
  cvt(wo,   wob,   (long long)IDc * HD);
  cvt(mask, maskb, (long long)Sc * Sc);

  { // G1: dkv = x @ Wdkv^T + b            (4096 x 576 x 4096)
    Gemm g = mkg(); g.M = BS; g.N = RR; g.K = IDc;
    g.A = xb; g.a_row = IDc; g.B = wdkvb; g.b_row = IDc;
    g.Cf = dkv; g.c_row = RR; g.bias = bdkv; run(g, 1);
  }
  { // G2: dq = x @ Wdq^T + b              (4096 x 512 x 4096)
    Gemm g = mkg(); g.M = BS; g.N = Rc; g.K = IDc;
    g.A = xb; g.a_row = IDc; g.B = wdqb; g.b_row = IDc;
    g.Cf = dq; g.c_row = Rc; g.bias = bdq; run(g, 1);
  }
  norm_rope<<<dim3(BS), dim3(256), 0, stream>>>(dkv, dq, kvnw, qnw, ch, sh, ckvkrT, cq);
  { // G3: q = c_q @ Wuq^T + b             (4096 x 4096 x 512), bf16 out
    Gemm g = mkg(); g.M = BS; g.N = HD; g.K = Rc;
    g.A = cq; g.a_row = Rc; g.B = wuqb; g.b_row = Rc;
    g.Cb = qb; g.c_row = HD; g.bias = buq; run(g, 1);
  }
  { // G4: qr0 = c_q @ Wqr^T + b           (4096 x 64 x 512)
    Gemm g = mkg(); g.M = BS; g.N = RPc; g.K = Rc;
    g.A = cq; g.a_row = Rc; g.B = wqrb; g.b_row = Rc;
    g.Cf = qr0; g.c_row = RPc; g.bias = bqr; run(g, 1);
  }
  rope64<<<dim3(BS), dim3(RPc), 0, stream>>>(qr0, ch, sh, qrb);
  { // G6: KCt[b][r][j] = sum_t ckvT[b][r][t]*kcatT[b][j][t]   (512 x 576 x 1024, batch B)
    Gemm g = mkg(); g.M = Rc; g.N = RR; g.K = Sc;
    g.A = ckvkrT; g.a_row = Sc; g.a_lo = (long long)RR * Sc;
    g.B = ckvkrT; g.b_row = Sc; g.b_lo = (long long)RR * Sc;
    g.Cb = KCt; g.c_row = RR; g.c_blo = (long long)Rc * RR; run(g, Bc);
  }
  { // G7: MC[b] = mask @ C_kv[b]          (1024 x 512 x 1024, batch B)
    Gemm g = mkg(); g.M = Sc; g.N = Rc; g.K = Sc;
    g.A = maskb; g.a_row = Sc;                       // same mask for all b
    g.B = ckvkrT; g.b_row = Sc; g.b_lo = (long long)RR * Sc;
    g.Cf = MC; g.c_row = Rc; g.c_blo = (long long)Sc * Rc; run(g, Bc);
  }
  { // GP: MC[b] += q_r @ KCt_pe[b]^T      (1024 x 512 x 64, batch B)
    Gemm g = mkg(); g.M = Sc; g.N = Rc; g.K = RPc;
    g.A = qrb; g.a_row = RPc; g.a_lo = (long long)Sc * RPc;
    g.B = KCt + Rc; g.b_row = RR; g.b_lo = (long long)Rc * RR; // cols 512..575
    g.Cf = MC; g.c_row = Rc; g.c_blo = (long long)Sc * Rc;
    g.add = MC; g.add_row = Rc; g.add_blo = (long long)Sc * Rc; run(g, Bc);
  }
  { // W2[b,h][r][d] = KCt[b] @ wuk_h      (512 x 128 x 512, batch B*H)
    Gemm g = mkg(); g.M = Rc; g.N = Dc; g.K = Rc;
    g.A = KCt; g.a_row = RR; g.a_sh = 5; g.a_hi = (long long)Rc * RR;
    g.B = wukb; g.b_row = Rc; g.b_sh = 5; g.b_lo = (long long)Dc * Rc;
    g.Cb = W2; g.c_row = Dc; g.c_blo = (long long)Rc * Dc; run(g, Bc * Hc);
  }
  { // G8: a_sum[b,h] = q[b,:,h,:] @ W2[b,h]^T + MC[b]   (1024 x 512 x 128, batch B*H)
    Gemm g = mkg(); g.M = Sc; g.N = Rc; g.K = Dc;
    g.A = qb; g.a_row = HD; g.a_sh = 5; g.a_hi = (long long)Sc * HD; g.a_lo = Dc;
    g.B = W2; g.b_row = Dc; g.b_lo = (long long)Rc * Dc;
    g.Cb = asum; g.c_row = Rc; g.c_blo = (long long)Sc * Rc;
    g.add = MC; g.add_row = Rc; g.add_bsh = 5; g.add_bhi = (long long)Sc * Rc;
    run(g, Bc * Hc);
  }
  { // G9: attn[b,s,h*128+d] = a_sum[b,h] @ wuv_h^T      (1024 x 128 x 512, batch B*H)
    Gemm g = mkg(); g.M = Sc; g.N = Dc; g.K = Rc;
    g.A = asum; g.a_row = Rc; g.a_lo = (long long)Sc * Rc;
    g.B = wuvb; g.b_row = Rc; g.b_sh = 5; g.b_lo = (long long)Dc * Rc;
    g.Cb = attnb; g.c_row = HD;
    g.c_bsh = 5; g.c_bhi = (long long)Sc * HD; g.c_blo = Dc; run(g, Bc * Hc);
  }
  { // G10: out = attn @ Wo^T + b          (4096 x 4096 x 4096)
    Gemm g = mkg(); g.M = BS; g.N = IDc; g.K = HD;
    g.A = attnb; g.a_row = HD; g.B = wob; g.b_row = HD;
    g.Cf = out; g.c_row = IDc; g.bias = bo; run(g, 1);
  }
}
